// RecurrentLayer_14482629722990
// MI455X (gfx1250) — matrix operand order repaired and checked by reading, not compiled, not run on hardware
//
#include <hip/hip_runtime.h>

// ---------------------------------------------------------------------------
// MI455X (gfx1250, wave32) gated-recurrent block.
//  - All 5 GEMMs: v_wmma_f32_16x16x32_bf16, fp32 accumulate.
//  - ALL GEMM A/B operands pre-materialized in bf16 by producer kernels:
//    inner loops are pure b128 loads + wmma (no per-iteration cvt VALU).
//  - gemm_gates: dual accumulators (rec+inp tiles) -> alpha/beta' fused.
//  - Recurrence: 3-phase chunked scan; phase C fuses silu(z)*h -> bf16 A.
//  - ffn1 hidden activation stored bf16-only (GEMM-A consumer only).
// ---------------------------------------------------------------------------

#define DM      256
#define HID     512
#define BN      32
#define LSEQ    2048
#define NROWS   (BN * LSEQ)        // 65536
#define KS      4
#define NCHUNK  16
#define CLEN    (LSEQ / NCHUNK)    // 128

typedef __attribute__((ext_vector_type(16))) __bf16 v16bf;
typedef __attribute__((ext_vector_type(8)))  __bf16 v8bf;
typedef __attribute__((ext_vector_type(8)))  float  v8f;

__device__ __forceinline__ float sigmoidf_(float x) { return 1.0f / (1.0f + __expf(-x)); }
__device__ __forceinline__ float siluf_(float x)    { return x * sigmoidf_(x); }

// fp32 -> bf16 round-to-nearest-even (integer ops; used OUTSIDE hot loops only)
__device__ __forceinline__ __bf16 f2bf(float f) {
    union { float f; unsigned u; } v; v.f = f;
    unsigned r = v.u + 0x7FFFu + ((v.u >> 16) & 1u);
    union { unsigned short s; __bf16 b; } o; o.s = (unsigned short)(r >> 16);
    return o.b;
}

// A fragment (16x32 bf16, contiguous k order): per lane one 16-elem run,
// {k0+grp*16 .. +15}. Two global_load_b128, zero VALU.
__device__ __forceinline__ v16bf load_a_bf(const __bf16* __restrict__ arow, int k0, int grp) {
    v8bf lo = *(const v8bf*)(arow + k0 + grp * 16);
    v8bf hi = *(const v8bf*)(arow + k0 + grp * 16 + 8);
    return __builtin_shufflevector(lo, hi, 0, 1, 2, 3, 4, 5, 6, 7,
                                   8, 9, 10, 11, 12, 13, 14, 15);
}

// ------------------------ weight convert + transpose -----------------------
// src: [K,N] fp32 -> dst: [N,K] bf16 (B fragments contiguous per lane)
__global__ void wt_convert(const float* __restrict__ W, __bf16* __restrict__ Wt, int K, int N) {
    int idx = blockIdx.x * 256 + threadIdx.x;
    if (idx >= K * N) return;
    int n = idx / K, k = idx - n * K;
    Wt[idx] = f2bf(W[(size_t)k * N + n]);
}

// ------------------------- fp32 -> bf16 stream (x4) ------------------------
__global__ void cvt_bf16x4(const float* __restrict__ src, __bf16* __restrict__ dst) {
    size_t i = ((size_t)blockIdx.x * 256 + threadIdx.x) * 4;
    float4 v = *(const float4*)(src + i);
    dst[i + 0] = f2bf(v.x); dst[i + 1] = f2bf(v.y);
    dst[i + 2] = f2bf(v.z); dst[i + 3] = f2bf(v.w);
}

// ------------------------------ generic GEMM -------------------------------
// A is bf16 row-major [M,K]. EPI: 0 = store f32 raw; 1 = +bias,silu -> bf16;
// 2 = +bias,+res -> f32; 3 = +res -> f32
template <int EPI>
__global__ void gemm_abf(const __bf16* __restrict__ A, const __bf16* __restrict__ Wt,
                         float* __restrict__ Cf, __bf16* __restrict__ Cb,
                         const float* __restrict__ bias, const float* __restrict__ res,
                         int K, int N) {
    const int nColBlk = N / 128;
    int bx    = blockIdx.x;
    int tileM = bx / nColBlk;
    int colB  = bx - tileM * nColBlk;
    int wave  = threadIdx.x >> 5;
    int lane  = threadIdx.x & 31;
    int row0  = tileM * 16;
    int colN  = colB * 128 + wave * 16;
    int mrow  = lane & 15;
    int grp   = lane >> 4;

    const __bf16* arow = A  + (size_t)(row0 + mrow) * K;
    const __bf16* wcol = Wt + (size_t)(colN + mrow) * K + grp * 16;

    v8f acc = {};
#pragma unroll 4
    for (int k0 = 0; k0 < K; k0 += 32) {
        v16bf a = load_a_bf(arow, k0, grp);
        v16bf b = *(const v16bf*)(wcol + k0);
        acc = __builtin_amdgcn_wmma_f32_16x16x32_bf16(false, a, false, b,
                                                      (short)0, acc, false, false);
    }

    int   c    = colN + mrow;
    float bval = (EPI == 1 || EPI == 2) ? bias[c] : 0.0f;
#pragma unroll
    for (int r = 0; r < 8; ++r) {
        int   m = row0 + r + 8 * grp;
        float v = acc[r] + bval;
        if (EPI == 1) {
            Cb[(size_t)m * N + c] = f2bf(siluf_(v));
        } else {
            if (EPI == 2 || EPI == 3) v += res[(size_t)m * N + c];
            Cf[(size_t)m * N + c] = v;
        }
    }
}

// -------------------- fused gate GEMM -> alpha, beta' ----------------------
__global__ void gemm_gates(const __bf16* __restrict__ xcb, const __bf16* __restrict__ Wt,
                           const float* __restrict__ bg, const float* __restrict__ Lam,
                           const float* __restrict__ xcf,
                           float* __restrict__ alpha, float* __restrict__ betap) {
    const int nColBlk = HID / 128;  // 4
    int bx    = blockIdx.x;
    int tileM = bx / nColBlk;
    int colB  = bx - tileM * nColBlk;
    int wave  = threadIdx.x >> 5;
    int lane  = threadIdx.x & 31;
    int row0  = tileM * 16;
    int colN  = colB * 128 + wave * 16;
    int mrow  = lane & 15;
    int grp   = lane >> 4;
    int c     = colN + mrow;

    const __bf16* arow = xcb + (size_t)(row0 + mrow) * HID;
    const __bf16* wR   = Wt + (size_t)c * HID + grp * 16;          // rec column
    const __bf16* wI   = Wt + (size_t)(c + 512) * HID + grp * 16;  // inp column

    v8f accR = {}, accI = {};
#pragma unroll 2
    for (int k0 = 0; k0 < HID; k0 += 32) {
        v16bf a  = load_a_bf(arow, k0, grp);
        v16bf bR = *(const v16bf*)(wR + k0);
        v16bf bI = *(const v16bf*)(wI + k0);
        accR = __builtin_amdgcn_wmma_f32_16x16x32_bf16(false, a, false, bR, (short)0, accR, false, false);
        accI = __builtin_amdgcn_wmma_f32_16x16x32_bf16(false, a, false, bI, (short)0, accI, false, false);
    }

    float sp   = log1pf(__expf(Lam[c]));  // softplus(Lambda[c]) — per-lane constant
    float brec = bg[c], binp = bg[c + 512];
#pragma unroll
    for (int r = 0; r < 8; ++r) {
        int    m   = row0 + r + 8 * grp;
        float  rec = accR[r] + brec;
        float  inp = accI[r] + binp;
        float  al  = __expf(-sp * sigmoidf_(rec));
        float  be  = sqrtf(fmaxf(1.0f - al * al + 1e-8f, 0.0f)) * sigmoidf_(inp);
        size_t idx = (size_t)m * HID + c;
        alpha[idx] = al;
        betap[idx] = be * xcf[idx];
    }
}

// ---------------------- depthwise causal conv + SiLU -----------------------
// Emits xc in fp32 (gate epilogue) AND bf16 (GEMM-A operand).
__global__ void dwconv(const float* __restrict__ xz, const float* __restrict__ cw,
                       const float* __restrict__ cb, float* __restrict__ xcf,
                       __bf16* __restrict__ xcb) {
    int rh  = blockIdx.x;
    int row = rh >> 1;
    int c   = ((rh & 1) << 8) + threadIdx.x;
    int t   = row & (LSEQ - 1);
    int br  = row - t;
    float acc = cb[c];
#pragma unroll
    for (int w = 0; w < KS; ++w) {
        int tt = t + w - (KS - 1);
        if (tt >= 0) acc = fmaf(cw[c * KS + w], xz[(size_t)(br + tt) * 1024 + c], acc);
    }
    float v = siluf_(acc);
    xcf[(size_t)row * HID + c] = v;
    xcb[(size_t)row * HID + c] = f2bf(v);
}

// ------------------------- 3-phase chunked scan ----------------------------
__global__ void scanA(const float* __restrict__ al, const float* __restrict__ bp,
                      float* __restrict__ sA, float* __restrict__ sB) {
    int blk = blockIdx.x, half = blk & 1, bj = blk >> 1;
    int j = bj % NCHUNK, b = bj / NCHUNK;
    int c = (half << 8) + threadIdx.x;
    size_t base = ((size_t)b * LSEQ + (size_t)j * CLEN) * HID + c;
    float A_ = 1.0f, B_ = 0.0f;
#pragma unroll 4
    for (int t = 0; t < CLEN; ++t) {
        float a = al[base + (size_t)t * HID];
        float x = bp[base + (size_t)t * HID];
        A_ *= a;
        B_ = fmaf(a, B_, x);
    }
    size_t s = ((size_t)b * HID + c) * NCHUNK + j;
    sA[s] = A_; sB[s] = B_;
}

__global__ void scanB(const float* __restrict__ sA, const float* __restrict__ sB,
                      float* __restrict__ carry) {
    int    i    = blockIdx.x * 256 + threadIdx.x;  // b*HID + c
    size_t base = (size_t)i * NCHUNK;
    float  h    = 0.0f;
#pragma unroll
    for (int j = 0; j < NCHUNK; ++j) {
        carry[base + j] = h;
        h = fmaf(sA[base + j], h, sB[base + j]);
    }
}

// Phase C fuses h-reconstruction with silu(z)*h and emits the bf16 A operand
// of the output projection directly (the fp32 h tensor never exists).
__global__ void scanC(const float* __restrict__ al, const float* __restrict__ bp,
                      const float* __restrict__ carry, const float* __restrict__ xz,
                      __bf16* __restrict__ azh) {
    int blk = blockIdx.x, half = blk & 1, bj = blk >> 1;
    int j = bj % NCHUNK, b = bj / NCHUNK;
    int c = (half << 8) + threadIdx.x;
    int row0 = b * LSEQ + j * CLEN;
    size_t base = (size_t)row0 * HID + c;
    float  h    = carry[((size_t)b * HID + c) * NCHUNK + j];
#pragma unroll 4
    for (int t = 0; t < CLEN; ++t) {
        float a = al[base + (size_t)t * HID];
        float x = bp[base + (size_t)t * HID];
        h = fmaf(a, h, x);
        float z = xz[(size_t)(row0 + t) * 1024 + 512 + c];
        azh[base + (size_t)t * HID] = f2bf(siluf_(z) * h);
    }
}

// ------------------------------- LayerNorm ---------------------------------
// One wave per 256-element row; EMITBF also writes a bf16 copy (GEMM-A feed).
template <int EMITBF>
__global__ void ln_kernel(const float* __restrict__ X, const float* __restrict__ g,
                          const float* __restrict__ bt, float* __restrict__ Y,
                          __bf16* __restrict__ Yb) {
    int row  = blockIdx.x * 8 + (threadIdx.x >> 5);
    int lane = threadIdx.x & 31;
    const float4* p = (const float4*)(X + (size_t)row * DM + lane * 8);
    float4 u0 = p[0], u1 = p[1];
    float vals[8] = {u0.x, u0.y, u0.z, u0.w, u1.x, u1.y, u1.z, u1.w};
    float s = 0.0f, sq = 0.0f;
#pragma unroll
    for (int j = 0; j < 8; ++j) { s += vals[j]; sq = fmaf(vals[j], vals[j], sq); }
#pragma unroll
    for (int off = 16; off > 0; off >>= 1) {
        s  += __shfl_xor(s, off, 32);
        sq += __shfl_xor(sq, off, 32);
    }
    float mean = s * (1.0f / DM);
    float var  = sq * (1.0f / DM) - mean * mean;
    float rs   = rsqrtf(var + 1e-12f);
    int base = lane * 8;
#pragma unroll
    for (int j = 0; j < 8; ++j) {
        float o = (vals[j] - mean) * rs * g[base + j] + bt[base + j];
        Y[(size_t)row * DM + base + j] = o;
        if (EMITBF) Yb[(size_t)row * DM + base + j] = f2bf(o);
    }
}

// --------------------------------- launch ----------------------------------
extern "C" void kernel_launch(void* const* d_in, const int* in_sizes, int n_in,
                              void* d_out, int out_size, void* d_ws, size_t ws_size,
                              hipStream_t stream) {
    (void)in_sizes; (void)n_in; (void)out_size; (void)ws_size;
    const float* x    = (const float*)d_in[0];
    const float* w_in = (const float*)d_in[1];
    const float* cw   = (const float*)d_in[2];
    const float* cb   = (const float*)d_in[3];
    const float* w_g  = (const float*)d_in[4];
    const float* b_g  = (const float*)d_in[5];
    const float* Lam  = (const float*)d_in[6];
    const float* w_o  = (const float*)d_in[7];
    const float* ln1g = (const float*)d_in[8];
    const float* ln1b = (const float*)d_in[9];
    const float* fw1  = (const float*)d_in[10];
    const float* fb1  = (const float*)d_in[11];
    const float* fw2  = (const float*)d_in[12];
    const float* fb2  = (const float*)d_in[13];
    const float* ln2g = (const float*)d_in[14];
    const float* ln2b = (const float*)d_in[15];
    float* out = (float*)d_out;

    char*  ws  = (char*)d_ws;
    size_t off = 0;
    auto take = [&](size_t bytes) -> char* {
        char* p = ws + off;
        off = (off + bytes + 255) & ~(size_t)255;
        return p;
    };
    __bf16* wt_in  = (__bf16*)take((size_t)256  * 1024 * 2);
    __bf16* wt_g   = (__bf16*)take((size_t)512  * 1024 * 2);
    __bf16* wt_o   = (__bf16*)take((size_t)512  * 256  * 2);
    __bf16* wt_f1  = (__bf16*)take((size_t)256  * 1024 * 2);
    __bf16* wt_f2  = (__bf16*)take((size_t)1024 * 256  * 2);
    __bf16* x_bf   = (__bf16*)take((size_t)NROWS * DM * 2);    // bf16 copy of x
    __bf16* hs_bf  = (__bf16*)take((size_t)NROWS * DM * 2);    // bf16 copy of hs
    float*  xz     = (float*)take((size_t)NROWS * 1024 * 4);   // xz; later u_bf lives here
    float*  xcf    = (float*)take((size_t)NROWS * HID * 4);    // xc fp32
    __bf16* xcb    = (__bf16*)take((size_t)NROWS * HID * 2);   // xc bf16; later silu(z)*h bf16
    float*  abuf   = (float*)take((size_t)NROWS * HID * 4);    // alpha; later y; later v
    float*  bbuf   = (float*)take((size_t)NROWS * HID * 4);    // beta'; later hs fp32
    float*  sA     = (float*)take((size_t)BN * HID * NCHUNK * 4);
    float*  sB     = (float*)take((size_t)BN * HID * NCHUNK * 4);
    float*  carry  = (float*)take((size_t)BN * HID * NCHUNK * 4);
    __bf16* u_bf   = (__bf16*)xz;   // ffn1 hidden activation (bf16-only), reuses xz
    __bf16* azh    = xcb;           // silu(z)*h bf16, reuses xc_bf

    // 0) weights -> transposed bf16; x -> bf16
    wt_convert<<<(256 * 1024 + 255) / 256, 256, 0, stream>>>(w_in, wt_in, 256, 1024);
    wt_convert<<<(512 * 1024 + 255) / 256, 256, 0, stream>>>(w_g,  wt_g,  512, 1024);
    wt_convert<<<(512 * 256  + 255) / 256, 256, 0, stream>>>(w_o,  wt_o,  512, 256);
    wt_convert<<<(256 * 1024 + 255) / 256, 256, 0, stream>>>(fw1,  wt_f1, 256, 1024);
    wt_convert<<<(1024 * 256 + 255) / 256, 256, 0, stream>>>(fw2,  wt_f2, 1024, 256);
    cvt_bf16x4<<<(NROWS * DM / 4) / 256, 256, 0, stream>>>(x, x_bf);

    // 1) xz = x @ w_in
    gemm_abf<0><<<(NROWS / 16) * (1024 / 128), 256, 0, stream>>>(x_bf, wt_in, xz, nullptr, nullptr, nullptr, 256, 1024);
    // 2) depthwise causal conv + SiLU (fp32 + bf16 outputs)
    dwconv<<<NROWS * 2, 256, 0, stream>>>(xz, cw, cb, xcf, xcb);
    // 3) gates GEMM (dual tiles) fused -> alpha, beta'
    gemm_gates<<<(NROWS / 16) * (HID / 128), 256, 0, stream>>>(xcb, wt_g, b_g, Lam, xcf, abuf, bbuf);
    // 4) chunked linear-recurrence scan; phase C emits bf16 silu(z)*h
    scanA<<<BN * NCHUNK * 2, 256, 0, stream>>>(abuf, bbuf, sA, sB);
    scanB<<<(BN * HID) / 256, 256, 0, stream>>>(sA, sB, carry);
    scanC<<<BN * NCHUNK * 2, 256, 0, stream>>>(abuf, bbuf, carry, xz, azh);
    // 5) y = (silu(z)*h) @ w_out + x   (y reuses alpha buffer)
    gemm_abf<3><<<(NROWS / 16) * (DM / 128), 256, 0, stream>>>(azh, wt_o, abuf, nullptr, nullptr, x, 512, 256);
    // 6) hs = LN1(y)  (fp32 into bbuf for the ffn2 residual, bf16 feed for ffn1)
    ln_kernel<1><<<NROWS / 8, 256, 0, stream>>>(abuf, ln1g, ln1b, bbuf, hs_bf);
    // 7) u = silu(hs @ ffn_w1 + b1) -> bf16 only (reuses xz buffer)
    gemm_abf<1><<<(NROWS / 16) * (1024 / 128), 256, 0, stream>>>(hs_bf, wt_f1, nullptr, u_bf, fb1, nullptr, 256, 1024);
    // 8) v = u @ ffn_w2 + b2 + hs       (v reuses alpha buffer)
    gemm_abf<2><<<(NROWS / 16) * (DM / 128), 256, 0, stream>>>(u_bf, wt_f2, abuf, nullptr, fb2, bbuf, 1024, 256);
    // 9) out = LN2(v)
    ln_kernel<0><<<NROWS / 8, 256, 0, stream>>>(abuf, ln2g, ln2b, out, nullptr);
}
